// VisionBDH_84241488544241
// MI455X (gfx1250) — matrix operand order repaired and checked by reading, not compiled, not run on hardware
//
#include <hip/hip_runtime.h>
#include <hip/hip_bf16.h>
#include <math.h>

// ---------------- problem constants ----------------
#define BB      8       // batch
#define PD      256     // model dim D
#define NHEAD   4
#define NF      4096    // N per head
#define NLAYERS 6
#define NTOK    197
#define TPAD    224     // tokens padded to multiple of 32 (7 tiles of 32)
#define NCLS    1000
#define PK      768     // 3*16*16 patch vector
#define EPSL    1e-5f

typedef _Float16 half_t;
typedef __attribute__((ext_vector_type(16))) _Float16 v16h;
typedef __attribute__((ext_vector_type(8)))  _Float16 v8h;
typedef __attribute__((ext_vector_type(8)))  float    v8f;

// D = A(16x32) * B(32x16) + C, f16 in / f32 out
#define WMMA_F32(a,b,c) __builtin_amdgcn_wmma_f32_16x16x32_f16(false,(a),false,(b),(short)0,(c),false,false)

// ---- fragment loaders (match CDNA5 16-bit WMMA VGPR layouts) ----
// A 16x32 (MxK): lane<16 holds row M=lane, K {0..7,16..23}; lane>=16 holds
// row M=lane-16, K {8..15,24..31}.  A is row-major [M][K].
__device__ __forceinline__ v16h load_a(const half_t* A, int lda, int row, int k0, int kh) {
    const half_t* p = A + (size_t)row * lda + k0 + kh * 8;
    v8h lo = *(const v8h*)(p);
    v8h hi = *(const v8h*)(p + 16);
    v16h r;
#pragma unroll
    for (int i = 0; i < 8; ++i) { r[i] = lo[i]; r[i + 8] = hi[i]; }
    return r;
}
// B 32x16 (KxN): lane<16 holds column N=lane with K {0..7,16..23}; lane>=16 holds
// column N=lane-16 with K {8..15,24..31}.  BT is stored [col][K] (i.e. B transposed).
__device__ __forceinline__ v16h load_b(const half_t* BT, int ldb, int col, int k0, int kh) {
    const half_t* p = BT + (size_t)col * ldb + k0 + kh * 8;
    v8h lo = *(const v8h*)(p);
    v8h hi = *(const v8h*)(p + 16);
    v16h r;
#pragma unroll
    for (int i = 0; i < 8; ++i) { r[i] = lo[i]; r[i + 8] = hi[i]; }
    return r;
}

__device__ __forceinline__ float wsum(float v) {
#pragma unroll
    for (int off = 16; off > 0; off >>= 1) v += __shfl_xor(v, off, 32);
    return v;
}

// ---------------- weight prep ----------------
__global__ void k_transpose_f16(const float* __restrict__ src, half_t* __restrict__ dst,
                                int R, int C) {
    size_t idx = (size_t)blockIdx.x * blockDim.x + threadIdx.x;
    if (idx >= (size_t)R * C) return;
    int r = (int)(idx / C), c = (int)(idx % C);
    dst[(size_t)c * R + r] = (half_t)src[idx];
}
__global__ void k_convert_f16(const float* __restrict__ src, half_t* __restrict__ dst, size_t n) {
    size_t idx = (size_t)blockIdx.x * blockDim.x + threadIdx.x;
    if (idx < n) dst[idx] = (half_t)src[idx];
}

// ---------------- patch extraction ----------------
// pat16[b][t][e] : t=1..196 hold patch t-1, rows 0 and 197.. are zero.
__global__ void k_patches(const float* __restrict__ x, half_t* __restrict__ pt) {
    size_t idx = (size_t)blockIdx.x * blockDim.x + threadIdx.x;
    size_t total = (size_t)BB * TPAD * PK;
    if (idx >= total) return;
    int e = (int)(idx % PK);
    int t = (int)((idx / PK) % TPAD);
    int b = (int)(idx / ((size_t)PK * TPAD));
    float v = 0.f;
    if (t >= 1 && t <= 196) {
        int p  = t - 1;
        int gy = p / 14, gx = p % 14;
        int cc = e / 256, rr = (e % 256) / 16, cl = e % 16;
        v = x[(((size_t)b * 3 + cc) * 224 + gy * 16 + rr) * 224 + gx * 16 + cl];
    }
    pt[idx] = (half_t)v;
}

// ---------------- GEMM kernels (one wave -> 32x64 tile, 2 A frags x 4 B frags) ----------------
// patch tokens: (TPAD x 768) @ (768 -> 256 cols) -> tmp f32
__global__ void k_patch_gemm(const half_t* __restrict__ pat, const half_t* __restrict__ pw,
                             float* __restrict__ outp) {
    int b = blockIdx.z;
    int m0 = blockIdx.x * 32, n0 = blockIdx.y * 64;
    int lane = threadIdx.x & 31, kh = lane >> 4, l15 = lane & 15;
    const half_t* A = pat + (size_t)b * TPAD * PK;
    v8f acc[2][4] = {};
    for (int k0 = 0; k0 < PK; k0 += 32) {
        v16h a0 = load_a(A, PK, m0 + l15, k0, kh);
        v16h a1 = load_a(A, PK, m0 + 16 + l15, k0, kh);
#pragma unroll
        for (int j = 0; j < 4; ++j) {
            v16h bf = load_b(pw, PK, n0 + 16 * j + l15, k0, kh);
            acc[0][j] = WMMA_F32(a0, bf, acc[0][j]);
            acc[1][j] = WMMA_F32(a1, bf, acc[1][j]);
        }
    }
    float* op = outp + (size_t)b * TPAD * PD;
#pragma unroll
    for (int mi = 0; mi < 2; ++mi)
#pragma unroll
        for (int i = 0; i < 8; ++i) {
            int row = m0 + mi * 16 + kh * 8 + i;
#pragma unroll
            for (int j = 0; j < 4; ++j)
                op[(size_t)row * PD + n0 + 16 * j + l15] = acc[mi][j][i];
        }
}

// x_sparse = relu(h @ encoder[h]) : (TPAD x 256) @ (256 -> 4096) -> f16
__global__ void k_xsparse(const half_t* __restrict__ hmat, const half_t* __restrict__ encT,
                          half_t* __restrict__ xs) {
    int z = blockIdx.z; int b = z >> 2; int hd = z & 3;
    int m0 = blockIdx.x * 32, n0 = blockIdx.y * 64;
    int lane = threadIdx.x & 31, kh = lane >> 4, l15 = lane & 15;
    const half_t* A  = hmat + (size_t)b * TPAD * PD;
    const half_t* BT = encT + (size_t)hd * NF * PD;
    v8f acc[2][4] = {};
    for (int k0 = 0; k0 < PD; k0 += 32) {
        v16h a0 = load_a(A, PD, m0 + l15, k0, kh);
        v16h a1 = load_a(A, PD, m0 + 16 + l15, k0, kh);
#pragma unroll
        for (int j = 0; j < 4; ++j) {
            v16h bf = load_b(BT, PD, n0 + 16 * j + l15, k0, kh);
            acc[0][j] = WMMA_F32(a0, bf, acc[0][j]);
            acc[1][j] = WMMA_F32(a1, bf, acc[1][j]);
        }
    }
    half_t* op = xs + (size_t)z * TPAD * NF;
#pragma unroll
    for (int mi = 0; mi < 2; ++mi)
#pragma unroll
        for (int i = 0; i < 8; ++i) {
            int row = m0 + mi * 16 + kh * 8 + i;
#pragma unroll
            for (int j = 0; j < 4; ++j)
                op[(size_t)row * NF + n0 + 16 * j + l15] = (half_t)fmaxf(acc[mi][j][i], 0.f);
        }
}

// qr = rope(t * freq, x_sparse); freq[pair] = 2^(-32*pair/N)  (THETA = 2^16)
__global__ void k_rope(const half_t* __restrict__ xs, half_t* __restrict__ qr) {
    size_t idx = (size_t)blockIdx.x * blockDim.x + threadIdx.x;
    size_t total = (size_t)BB * NHEAD * TPAD * (NF / 2);
    if (idx >= total) return;
    int pair = (int)(idx % (NF / 2));
    int t    = (int)((idx / (NF / 2)) % TPAD);
    size_t base = (idx / (NF / 2)) * (size_t)NF + 2 * pair;
    float x0 = (float)xs[base], x1 = (float)xs[base + 1];
    float ph = (float)t * exp2f(-32.0f * (float)pair / (float)NF);
    float s = __sinf(ph), c = __cosf(ph);
    qr[base]     = (half_t)(x0 * c - x1 * s);
    qr[base + 1] = (half_t)(x1 * c + x0 * s);
}

// scores = mask(qr @ qr^T); row-major qr is already the [col][K] layout of qr^T
// wave tile: 32 rows x 16 cols (B fragment reused by both row tiles)
__global__ void k_scores(const half_t* __restrict__ qr, half_t* __restrict__ sc) {
    int z = blockIdx.z;
    int m0 = blockIdx.x * 32, n0 = blockIdx.y * 16;
    int lane = threadIdx.x & 31, kh = lane >> 4, l15 = lane & 15;
    const half_t* Q = qr + (size_t)z * TPAD * NF;
    v8f acc0 = {}, acc1 = {};
    for (int k0 = 0; k0 < NF; k0 += 32) {
        v16h a0 = load_a(Q, NF, m0 + l15, k0, kh);
        v16h a1 = load_a(Q, NF, m0 + 16 + l15, k0, kh);
        v16h bf = load_b(Q, NF, n0 + l15, k0, kh);
        acc0 = WMMA_F32(a0, bf, acc0);
        acc1 = WMMA_F32(a1, bf, acc1);
    }
    half_t* op = sc + (size_t)z * TPAD * TPAD;
    int col = n0 + l15;
#pragma unroll
    for (int i = 0; i < 8; ++i) {
        int r0 = m0 + kh * 8 + i;
        int r1 = r0 + 16;
        float v0 = (r0 > col && r0 < NTOK && col < NTOK) ? acc0[i] : 0.f;
        float v1 = (r1 > col && r1 < NTOK && col < NTOK) ? acc1[i] : 0.f;
        op[(size_t)r0 * TPAD + col] = (half_t)v0;
        op[(size_t)r1 * TPAD + col] = (half_t)v1;
    }
}

// yKV_raw = scores @ h : (TPAD x TPAD) @ (TPAD -> 256) -> f32
__global__ void k_ykv(const half_t* __restrict__ sc, const half_t* __restrict__ hT,
                      float* __restrict__ yk) {
    int z = blockIdx.z; int b = z >> 2;
    int m0 = blockIdx.x * 32, n0 = blockIdx.y * 64;
    int lane = threadIdx.x & 31, kh = lane >> 4, l15 = lane & 15;
    const half_t* A  = sc + (size_t)z * TPAD * TPAD;
    const half_t* BT = hT + (size_t)b * PD * TPAD;   // [d][t] layout
    v8f acc[2][4] = {};
    for (int k0 = 0; k0 < TPAD; k0 += 32) {
        v16h a0 = load_a(A, TPAD, m0 + l15, k0, kh);
        v16h a1 = load_a(A, TPAD, m0 + 16 + l15, k0, kh);
#pragma unroll
        for (int j = 0; j < 4; ++j) {
            v16h bf = load_b(BT, TPAD, n0 + 16 * j + l15, k0, kh);
            acc[0][j] = WMMA_F32(a0, bf, acc[0][j]);
            acc[1][j] = WMMA_F32(a1, bf, acc[1][j]);
        }
    }
    float* op = yk + (size_t)z * TPAD * PD;
#pragma unroll
    for (int mi = 0; mi < 2; ++mi)
#pragma unroll
        for (int i = 0; i < 8; ++i) {
            int row = m0 + mi * 16 + kh * 8 + i;
#pragma unroll
            for (int j = 0; j < 4; ++j)
                op[(size_t)row * PD + n0 + 16 * j + l15] = acc[mi][j][i];
        }
}

// xy = relu(yKV @ encoder_v[h]) * x_sparse -> f16 (written into qr buffer)
__global__ void k_ysparse(const half_t* __restrict__ yk16, const half_t* __restrict__ encvT,
                          const half_t* __restrict__ xs, half_t* __restrict__ xy) {
    int z = blockIdx.z; int hd = z & 3;
    int m0 = blockIdx.x * 32, n0 = blockIdx.y * 64;
    int lane = threadIdx.x & 31, kh = lane >> 4, l15 = lane & 15;
    const half_t* A  = yk16 + (size_t)z * TPAD * PD;
    const half_t* BT = encvT + (size_t)hd * NF * PD;
    v8f acc[2][4] = {};
    for (int k0 = 0; k0 < PD; k0 += 32) {
        v16h a0 = load_a(A, PD, m0 + l15, k0, kh);
        v16h a1 = load_a(A, PD, m0 + 16 + l15, k0, kh);
#pragma unroll
        for (int j = 0; j < 4; ++j) {
            v16h bf = load_b(BT, PD, n0 + 16 * j + l15, k0, kh);
            acc[0][j] = WMMA_F32(a0, bf, acc[0][j]);
            acc[1][j] = WMMA_F32(a1, bf, acc[1][j]);
        }
    }
    const half_t* xp = xs + (size_t)z * TPAD * NF;
    half_t* op = xy + (size_t)z * TPAD * NF;
#pragma unroll
    for (int mi = 0; mi < 2; ++mi)
#pragma unroll
        for (int i = 0; i < 8; ++i) {
            int row = m0 + mi * 16 + kh * 8 + i;
#pragma unroll
            for (int j = 0; j < 4; ++j) {
                size_t o = (size_t)row * NF + n0 + 16 * j + l15;
                float y = fmaxf(acc[mi][j][i], 0.f);
                op[o] = (half_t)(y * (float)xp[o]);
            }
        }
}

// yMLP = sum_h xy[h] @ decoder[h] : K = 4 * 4096 -> f32
__global__ void k_ymlp(const half_t* __restrict__ xy, const half_t* __restrict__ decT,
                       float* __restrict__ ym) {
    int b = blockIdx.z;
    int m0 = blockIdx.x * 32, n0 = blockIdx.y * 64;
    int lane = threadIdx.x & 31, kh = lane >> 4, l15 = lane & 15;
    v8f acc[2][4] = {};
    for (int hd = 0; hd < NHEAD; ++hd) {
        const half_t* A  = xy + ((size_t)(b * NHEAD + hd)) * TPAD * NF;
        const half_t* BT = decT + (size_t)hd * PD * NF;   // [d][n]
        for (int k0 = 0; k0 < NF; k0 += 32) {
            v16h a0 = load_a(A, NF, m0 + l15, k0, kh);
            v16h a1 = load_a(A, NF, m0 + 16 + l15, k0, kh);
#pragma unroll
            for (int j = 0; j < 4; ++j) {
                v16h bf = load_b(BT, NF, n0 + 16 * j + l15, k0, kh);
                acc[0][j] = WMMA_F32(a0, bf, acc[0][j]);
                acc[1][j] = WMMA_F32(a1, bf, acc[1][j]);
            }
        }
    }
    float* op = ym + (size_t)b * TPAD * PD;
#pragma unroll
    for (int mi = 0; mi < 2; ++mi)
#pragma unroll
        for (int i = 0; i < 8; ++i) {
            int row = m0 + mi * 16 + kh * 8 + i;
#pragma unroll
            for (int j = 0; j < 4; ++j)
                op[(size_t)row * PD + n0 + 16 * j + l15] = acc[mi][j][i];
        }
}

// ---------------- LayerNorm kernels (one wave32 per 256-wide row) ----------------
__global__ void k_ln_seq(const float* __restrict__ seq, const float* __restrict__ pb,
                         const float* __restrict__ cls, const float* __restrict__ pos,
                         float* __restrict__ h32, half_t* __restrict__ h16,
                         half_t* __restrict__ hT) {
    int row = blockIdx.x;                 // b*TPAD + t
    int b = row / TPAD, t = row % TPAD;
    int lane = threadIdx.x;
    float x[8];
#pragma unroll
    for (int i = 0; i < 8; ++i) {
        int d = lane * 8 + i;
        float v;
        if (t == 0)          v = cls[d] + pos[d];
        else if (t < NTOK)   v = seq[(size_t)row * PD + d] + pb[d] + pos[(size_t)t * PD + d];
        else                 v = 0.f;
        x[i] = v;
    }
    float s = 0.f;
#pragma unroll
    for (int i = 0; i < 8; ++i) s += x[i];
    float m = wsum(s) / PD;
    float vs = 0.f;
#pragma unroll
    for (int i = 0; i < 8; ++i) { float d0 = x[i] - m; vs += d0 * d0; }
    float rs = rsqrtf(wsum(vs) / PD + EPSL);
#pragma unroll
    for (int i = 0; i < 8; ++i) {
        int d = lane * 8 + i;
        float o = (x[i] - m) * rs;
        h32[(size_t)row * PD + d] = o;
        h16[(size_t)row * PD + d] = (half_t)o;
        hT[((size_t)b * PD + d) * TPAD + t] = (half_t)o;
    }
}

__global__ void k_ln_ykv(const float* __restrict__ in, half_t* __restrict__ op) {
    int row = blockIdx.x;
    int lane = threadIdx.x;
    float x[8];
#pragma unroll
    for (int i = 0; i < 8; ++i) x[i] = in[(size_t)row * PD + lane * 8 + i];
    float s = 0.f;
#pragma unroll
    for (int i = 0; i < 8; ++i) s += x[i];
    float m = wsum(s) / PD;
    float vs = 0.f;
#pragma unroll
    for (int i = 0; i < 8; ++i) { float d0 = x[i] - m; vs += d0 * d0; }
    float rs = rsqrtf(wsum(vs) / PD + EPSL);
#pragma unroll
    for (int i = 0; i < 8; ++i)
        op[(size_t)row * PD + lane * 8 + i] = (half_t)((x[i] - m) * rs);
}

// h = ln(h + ln(yMLP))
__global__ void k_final_ln(const float* __restrict__ ym, float* __restrict__ h32,
                           half_t* __restrict__ h16, half_t* __restrict__ hT) {
    int row = blockIdx.x;
    int b = row / TPAD, t = row % TPAD;
    int lane = threadIdx.x;
    float y[8];
#pragma unroll
    for (int i = 0; i < 8; ++i) y[i] = ym[(size_t)row * PD + lane * 8 + i];
    float s = 0.f;
#pragma unroll
    for (int i = 0; i < 8; ++i) s += y[i];
    float m1 = wsum(s) / PD;
    float vs = 0.f;
#pragma unroll
    for (int i = 0; i < 8; ++i) { float d0 = y[i] - m1; vs += d0 * d0; }
    float rs1 = rsqrtf(wsum(vs) / PD + EPSL);
    float x[8];
#pragma unroll
    for (int i = 0; i < 8; ++i)
        x[i] = h32[(size_t)row * PD + lane * 8 + i] + (y[i] - m1) * rs1;
    s = 0.f;
#pragma unroll
    for (int i = 0; i < 8; ++i) s += x[i];
    float m2 = wsum(s) / PD;
    vs = 0.f;
#pragma unroll
    for (int i = 0; i < 8; ++i) { float d0 = x[i] - m2; vs += d0 * d0; }
    float rs2 = rsqrtf(wsum(vs) / PD + EPSL);
#pragma unroll
    for (int i = 0; i < 8; ++i) {
        int d = lane * 8 + i;
        float o = (x[i] - m2) * rs2;
        h32[(size_t)row * PD + d] = o;
        h16[(size_t)row * PD + d] = (half_t)o;
        hT[((size_t)b * PD + d) * TPAD + t] = (half_t)o;
    }
}

// ---------------- head ----------------
__global__ void k_cls_ln(const float* __restrict__ h32, const float* __restrict__ g,
                         const float* __restrict__ bvec, float* __restrict__ clsn) {
    int b = blockIdx.x;
    int lane = threadIdx.x;
    const float* p = h32 + (size_t)b * TPAD * PD;   // token 0
    float x[8];
#pragma unroll
    for (int i = 0; i < 8; ++i) x[i] = p[lane * 8 + i];
    float s = 0.f;
#pragma unroll
    for (int i = 0; i < 8; ++i) s += x[i];
    float m = wsum(s) / PD;
    float vs = 0.f;
#pragma unroll
    for (int i = 0; i < 8; ++i) { float d0 = x[i] - m; vs += d0 * d0; }
    float rs = rsqrtf(wsum(vs) / PD + EPSL);
#pragma unroll
    for (int i = 0; i < 8; ++i) {
        int d = lane * 8 + i;
        clsn[(size_t)b * PD + d] = (x[i] - m) * rs * g[d] + bvec[d];
    }
}

__global__ void k_head(const float* __restrict__ clsn, const float* __restrict__ W,
                       const float* __restrict__ bias, float* __restrict__ out) {
    int idx = blockIdx.x * blockDim.x + threadIdx.x;
    if (idx >= BB * NCLS) return;
    int b = idx / NCLS, j = idx % NCLS;
    float acc = bias[j];
    for (int k = 0; k < PD; ++k)
        acc += clsn[(size_t)b * PD + k] * W[(size_t)k * NCLS + j];
    out[idx] = acc;
}

// ---------------- host ----------------
extern "C" void kernel_launch(void* const* d_in, const int* in_sizes, int n_in,
                              void* d_out, int out_size, void* d_ws, size_t ws_size,
                              hipStream_t stream) {
    (void)in_sizes; (void)n_in; (void)out_size; (void)ws_size;
    const float* x         = (const float*)d_in[0];
    const float* patch_w   = (const float*)d_in[1];
    const float* patch_b   = (const float*)d_in[2];
    const float* cls_tok   = (const float*)d_in[3];
    const float* pos_emb   = (const float*)d_in[4];
    const float* encoder   = (const float*)d_in[5];
    const float* encoder_v = (const float*)d_in[6];
    const float* decoder   = (const float*)d_in[7];
    const float* head_g    = (const float*)d_in[8];
    const float* head_b    = (const float*)d_in[9];
    const float* head_w    = (const float*)d_in[10];
    const float* head_bias = (const float*)d_in[11];
    float* out = (float*)d_out;

    char* ws = (char*)d_ws;
    size_t off = 0;
    auto alloc = [&](size_t bytes) -> void* {
        void* p = ws + off;
        off = (off + bytes + 255) & ~(size_t)255;
        return p;
    };
    float*  tmp_td = (float*) alloc((size_t)BB * TPAD * PD * 4);
    float*  h32    = (float*) alloc((size_t)BB * TPAD * PD * 4);
    half_t* h16    = (half_t*)alloc((size_t)BB * TPAD * PD * 2);
    half_t* hT     = (half_t*)alloc((size_t)BB * PD * TPAD * 2);
    half_t* encT   = (half_t*)alloc((size_t)NHEAD * NF * PD * 2);
    half_t* encvT  = (half_t*)alloc((size_t)NHEAD * NF * PD * 2);
    half_t* decT   = (half_t*)alloc((size_t)NHEAD * PD * NF * 2);
    half_t* pw16   = (half_t*)alloc((size_t)PD * PK * 2);
    half_t* pat16  = (half_t*)alloc((size_t)BB * TPAD * PK * 2);
    half_t* xs     = (half_t*)alloc((size_t)BB * NHEAD * TPAD * NF * 2);
    half_t* xy     = (half_t*)alloc((size_t)BB * NHEAD * TPAD * NF * 2); // qr, then x*y
    half_t* sc     = (half_t*)alloc((size_t)BB * NHEAD * TPAD * TPAD * 2);
    float*  ykv32  = (float*) alloc((size_t)BB * NHEAD * TPAD * PD * 4);
    half_t* ykv16  = (half_t*)alloc((size_t)BB * NHEAD * TPAD * PD * 2);
    float*  clsn   = (float*) alloc((size_t)BB * PD * 4);

    // ---- weight prep (once per call; weights are shared across all layers) ----
    {
        int nel = PD * NF;
        int blocks = (nel + 255) / 256;
        for (int hd = 0; hd < NHEAD; ++hd) {
            k_transpose_f16<<<blocks, 256, 0, stream>>>(
                encoder + (size_t)hd * PD * NF, encT + (size_t)hd * NF * PD, PD, NF);
            k_transpose_f16<<<blocks, 256, 0, stream>>>(
                encoder_v + (size_t)hd * PD * NF, encvT + (size_t)hd * NF * PD, PD, NF);
            k_transpose_f16<<<blocks, 256, 0, stream>>>(
                decoder + (size_t)hd * NF * PD, decT + (size_t)hd * PD * NF, NF, PD);
        }
        size_t npw = (size_t)PD * PK;
        k_convert_f16<<<(int)((npw + 255) / 256), 256, 0, stream>>>(patch_w, pw16, npw);
    }

    // ---- patch embedding + ln ----
    {
        size_t npat = (size_t)BB * TPAD * PK;
        k_patches<<<(int)((npat + 255) / 256), 256, 0, stream>>>(x, pat16);
        k_patch_gemm<<<dim3(TPAD / 32, PD / 64, BB), 32, 0, stream>>>(pat16, pw16, tmp_td);
        k_ln_seq<<<BB * TPAD, 32, 0, stream>>>(tmp_td, patch_b, cls_tok, pos_emb, h32, h16, hT);
    }

    // ---- layers ----
    size_t nrope = (size_t)BB * NHEAD * TPAD * (NF / 2);
    for (int layer = 0; layer < NLAYERS; ++layer) {
        k_xsparse<<<dim3(TPAD / 32, NF / 64, BB * NHEAD), 32, 0, stream>>>(h16, encT, xs);
        k_rope<<<(int)((nrope + 255) / 256), 256, 0, stream>>>(xs, xy);              // xy := qr
        k_scores<<<dim3(TPAD / 32, TPAD / 16, BB * NHEAD), 32, 0, stream>>>(xy, sc);
        k_ykv<<<dim3(TPAD / 32, PD / 64, BB * NHEAD), 32, 0, stream>>>(sc, hT, ykv32);
        k_ln_ykv<<<BB * NHEAD * TPAD, 32, 0, stream>>>(ykv32, ykv16);
        k_ysparse<<<dim3(TPAD / 32, NF / 64, BB * NHEAD), 32, 0, stream>>>(ykv16, encvT, xs, xy);
        k_ymlp<<<dim3(TPAD / 32, PD / 64, BB), 32, 0, stream>>>(xy, decT, tmp_td);
        k_final_ln<<<BB * TPAD, 32, 0, stream>>>(tmp_td, h32, h16, hT);
    }

    // ---- head ----
    k_cls_ln<<<BB, 32, 0, stream>>>(h32, head_g, head_b, clsn);
    k_head<<<(BB * NCLS + 255) / 256, 256, 0, stream>>>(clsn, head_w, head_bias, out);
}
